// GATNet_54022098649653
// MI455X (gfx1250) — matrix operand order repaired and checked by reading, not compiled, not run on hardware
//
#include <hip/hip_runtime.h>

// GAT 2-layer fused pipeline for MI455X (gfx1250, wave32).
// Dense node transforms use v_wmma_f32_16x16x32_f16 (16-node tiles per wave).
// Edge softmax: atomic-max (ordered-uint trick) + unnormalized atomic-add
// accumulation + late division; self-loops folded in analytically.
// All node-state scratch (~35 MB) is L2-resident (192 MB L2).

typedef __attribute__((ext_vector_type(16))) _Float16 v16h;
typedef __attribute__((ext_vector_type(8)))  float    v8f;

#define NEG_SLOPE 0.2f

__device__ __forceinline__ unsigned fenc(float f) {
  unsigned u = __float_as_uint(f);
  return (u & 0x80000000u) ? ~u : (u | 0x80000000u);
}
__device__ __forceinline__ float fdec(unsigned k) {
  unsigned u = (k & 0x80000000u) ? (k ^ 0x80000000u) : ~k;
  return __uint_as_float(u);
}
__device__ __forceinline__ float lrelu(float v) {
  return v > 0.f ? v : NEG_SLOPE * v;
}

// ---------------- Layer 1: h1 = x @ W1  (K=2 padded into 16x16x32 WMMA) ----
__global__ void k_transform1(const float* __restrict__ x,
                             const float* __restrict__ W1,
                             float* __restrict__ h1, int N) {
  int wave = (blockIdx.x * blockDim.x + threadIdx.x) >> 5;
  int lane = threadIdx.x & 31;
  int tiles = (N + 15) >> 4;
  if (wave >= tiles) return;               // wave-uniform guard (EXEC stays all-1)

  int row  = lane & 15;
  int base = wave << 4;
  int node = base + row;

  // A fragment (16x32 f16): only K=0,1 nonzero -> lanes 0..15, elems 0,1.
  v16h a = {};
  if (lane < 16 && node < N) {
    a[0] = (_Float16)x[node * 2 + 0];
    a[1] = (_Float16)x[node * 2 + 1];
  }
  // B fragments (32x16 f16), cols 0..15 and 16..31 of W1; only K=0,1 nonzero.
  v16h b0 = {}, b1v = {};
  if (lane < 16) {
    int col = lane;
    b0[0]  = (_Float16)W1[0 * 32 + col];
    b0[1]  = (_Float16)W1[1 * 32 + col];
    b1v[0] = (_Float16)W1[0 * 32 + col + 16];
    b1v[1] = (_Float16)W1[1 * 32 + col + 16];
  }
  v8f c0 = {}, c1 = {};
  c0 = __builtin_amdgcn_wmma_f32_16x16x32_f16(false, a, false, b0,  (short)0, c0, false, false);
  c1 = __builtin_amdgcn_wmma_f32_16x16x32_f16(false, a, false, b1v, (short)0, c1, false, false);

  // D layout: lanes 0..15 -> N=lane, M=r ; lanes 16..31 -> N=lane-16, M=8+r
  int col  = lane & 15;
  int mofs = (lane < 16) ? 0 : 8;
  if (base + 16 <= N) {                    // wave-uniform: full tile, no guards
#pragma unroll
    for (int r = 0; r < 8; ++r) {
      int n2 = base + r + mofs;
      h1[n2 * 32 + col]      = c0[r];
      h1[n2 * 32 + col + 16] = c1[r];
    }
  } else {
#pragma unroll
    for (int r = 0; r < 8; ++r) {
      int n2 = base + r + mofs;
      if (n2 < N) {
        h1[n2 * 32 + col]      = c0[r];
        h1[n2 * 32 + col + 16] = c1[r];
      }
    }
  }
}

// logits + per-layer init: m1 <- enc(self-loop logit), den1/acc1 <- 0
__global__ void k_logits1(const float* __restrict__ h1,
                          const float* __restrict__ as1,
                          const float* __restrict__ ad1,
                          float* __restrict__ als, float* __restrict__ ald,
                          unsigned* __restrict__ m1, float* __restrict__ den1,
                          float* __restrict__ acc1, int N) {
  int i = blockIdx.x * blockDim.x + threadIdx.x;
  if (i >= N) return;
  const float4* hv = (const float4*)(h1 + (size_t)i * 32);
  float4 z = make_float4(0.f, 0.f, 0.f, 0.f);
  float4* accv = (float4*)(acc1 + (size_t)i * 32);
#pragma unroll
  for (int h = 0; h < 4; ++h) {
    float4 p = hv[h * 2 + 0], q = hv[h * 2 + 1];
    const float* ash = as1 + h * 8;
    const float* adh = ad1 + h * 8;
    float s = p.x * ash[0] + p.y * ash[1] + p.z * ash[2] + p.w * ash[3]
            + q.x * ash[4] + q.y * ash[5] + q.z * ash[6] + q.w * ash[7];
    float d = p.x * adh[0] + p.y * adh[1] + p.z * adh[2] + p.w * adh[3]
            + q.x * adh[4] + q.y * adh[5] + q.z * adh[6] + q.w * adh[7];
    als[i * 4 + h] = s;
    ald[i * 4 + h] = d;
    m1[i * 4 + h]  = fenc(lrelu(s + d));   // self-loop seeds the running max
    den1[i * 4 + h] = 0.f;
    accv[h * 2 + 0] = z;
    accv[h * 2 + 1] = z;
  }
}

// Edge pass A: segment max (1 thread per edge, vector logit loads)
__global__ void k_edge_max1(const int* __restrict__ ei,
                            const float4* __restrict__ als,
                            const float4* __restrict__ ald,
                            unsigned* __restrict__ m1, int E) {
  int e = blockIdx.x * blockDim.x + threadIdx.x;
  if (e >= E) return;
  int s = ei[e], d = ei[E + e];
  float4 a = als[s], b = ald[d];
  atomicMax(&m1[d * 4 + 0], fenc(lrelu(a.x + b.x)));
  atomicMax(&m1[d * 4 + 1], fenc(lrelu(a.y + b.y)));
  atomicMax(&m1[d * 4 + 2], fenc(lrelu(a.z + b.z)));
  atomicMax(&m1[d * 4 + 3], fenc(lrelu(a.w + b.w)));
}

// Edge pass B: unnormalized exp-weighted accumulation (1 thread per edge).
// 36 non-returning f32 atomics per thread: fire-and-forget (STOREcnt).
__global__ void k_edge_acc1(const int* __restrict__ ei,
                            const float4* __restrict__ als,
                            const float4* __restrict__ ald,
                            const uint4* __restrict__ m1,
                            const float* __restrict__ h1,
                            float* __restrict__ den1, float* __restrict__ acc1,
                            int E) {
  int e = blockIdx.x * blockDim.x + threadIdx.x;
  if (e >= E) return;
  int s = ei[e], d = ei[E + e];
  float4 a = als[s], b = ald[d];
  uint4  mk = m1[d];
  float w0 = __expf(lrelu(a.x + b.x) - fdec(mk.x));
  float w1 = __expf(lrelu(a.y + b.y) - fdec(mk.y));
  float w2 = __expf(lrelu(a.z + b.z) - fdec(mk.z));
  float w3 = __expf(lrelu(a.w + b.w) - fdec(mk.w));
  atomicAdd(&den1[d * 4 + 0], w0);
  atomicAdd(&den1[d * 4 + 1], w1);
  atomicAdd(&den1[d * 4 + 2], w2);
  atomicAdd(&den1[d * 4 + 3], w3);
  const float4* hv = (const float4*)(h1 + (size_t)s * 32);
  float* accd = acc1 + (size_t)d * 32;
  float w[4] = {w0, w1, w2, w3};
#pragma unroll
  for (int h = 0; h < 4; ++h) {
    float4 p = hv[h * 2 + 0], q = hv[h * 2 + 1];
    float ww = w[h];
    atomicAdd(&accd[h * 8 + 0], ww * p.x);
    atomicAdd(&accd[h * 8 + 1], ww * p.y);
    atomicAdd(&accd[h * 8 + 2], ww * p.z);
    atomicAdd(&accd[h * 8 + 3], ww * p.w);
    atomicAdd(&accd[h * 8 + 4], ww * q.x);
    atomicAdd(&accd[h * 8 + 5], ww * q.y);
    atomicAdd(&accd[h * 8 + 6], ww * q.z);
    atomicAdd(&accd[h * 8 + 7], ww * q.w);
  }
}

// Finalize layer1: add self term, normalize, bias, ELU; write h2 over h1.
__global__ void k_finalize1(float* __restrict__ h1,
                            const float* __restrict__ als,
                            const float* __restrict__ ald,
                            const unsigned* __restrict__ m1,
                            const float* __restrict__ den1,
                            const float* __restrict__ acc1,
                            const float* __restrict__ b1, int N) {
  int t = blockIdx.x * blockDim.x + threadIdx.x;
  if (t >= N * 4) return;
  int h = t & 3, i = t >> 2;
  float es = lrelu(als[i * 4 + h] + ald[i * 4 + h]);
  float ws = __expf(es - fdec(m1[i * 4 + h]));        // self-loop weight
  float rden = 1.f / (den1[i * 4 + h] + ws + 1e-16f);
  float4* hv = (float4*)(h1 + (size_t)i * 32 + h * 8);
  const float4* av = (const float4*)(acc1 + (size_t)i * 32 + h * 8);
  const float4* bv = (const float4*)(b1 + h * 8);
  float4 hp = hv[0], hq = hv[1];
  float4 ap = av[0], aq = av[1];
  float4 bp = bv[0], bq = bv[1];
  float4 op, oq;
  op.x = (ap.x + ws * hp.x) * rden + bp.x;
  op.y = (ap.y + ws * hp.y) * rden + bp.y;
  op.z = (ap.z + ws * hp.z) * rden + bp.z;
  op.w = (ap.w + ws * hp.w) * rden + bp.w;
  oq.x = (aq.x + ws * hq.x) * rden + bq.x;
  oq.y = (aq.y + ws * hq.y) * rden + bq.y;
  oq.z = (aq.z + ws * hq.z) * rden + bq.z;
  oq.w = (aq.w + ws * hq.w) * rden + bq.w;
  op.x = op.x > 0.f ? op.x : (__expf(op.x) - 1.f);
  op.y = op.y > 0.f ? op.y : (__expf(op.y) - 1.f);
  op.z = op.z > 0.f ? op.z : (__expf(op.z) - 1.f);
  op.w = op.w > 0.f ? op.w : (__expf(op.w) - 1.f);
  oq.x = oq.x > 0.f ? oq.x : (__expf(oq.x) - 1.f);
  oq.y = oq.y > 0.f ? oq.y : (__expf(oq.y) - 1.f);
  oq.z = oq.z > 0.f ? oq.z : (__expf(oq.z) - 1.f);
  oq.w = oq.w > 0.f ? oq.w : (__expf(oq.w) - 1.f);
  hv[0] = op;
  hv[1] = oq;
}

// ---------------- Layer 2: t2 = h2 @ W2 (K=32 exact WMMA fit) --------------
__global__ void k_transform2(const float* __restrict__ h2,
                             const float* __restrict__ W2,
                             float* __restrict__ t2, int N) {
  int wave = (blockIdx.x * blockDim.x + threadIdx.x) >> 5;
  int lane = threadIdx.x & 31;
  int tiles = (N + 15) >> 4;
  if (wave >= tiles) return;

  int row  = lane & 15;
  int base = wave << 4;
  int node = base + row;
  // A fragment: lane<16 holds K0-7 / K16-23 ; lane>=16 holds K8-15 / K24-31
  v16h a = {};
  if (node < N) {
    int kb0 = (lane < 16) ? 0 : 8;
    int kb1 = (lane < 16) ? 16 : 24;
    const float4* hv = (const float4*)(h2 + (size_t)node * 32);
    float4 p0 = hv[kb0 >> 2], p1 = hv[(kb0 >> 2) + 1];
    float4 q0 = hv[kb1 >> 2], q1 = hv[(kb1 >> 2) + 1];
    a[0] = (_Float16)p0.x;  a[1] = (_Float16)p0.y;
    a[2] = (_Float16)p0.z;  a[3] = (_Float16)p0.w;
    a[4] = (_Float16)p1.x;  a[5] = (_Float16)p1.y;
    a[6] = (_Float16)p1.z;  a[7] = (_Float16)p1.w;
    a[8]  = (_Float16)q0.x; a[9]  = (_Float16)q0.y;
    a[10] = (_Float16)q0.z; a[11] = (_Float16)q0.w;
    a[12] = (_Float16)q1.x; a[13] = (_Float16)q1.y;
    a[14] = (_Float16)q1.z; a[15] = (_Float16)q1.w;
  }
  // B fragment: lane n<16 = col n with K0-7 / K16-23; lane n+16 = col n with K8-15 / K24-31.
  // Only cols 0,1 of W2 exist.
  v16h b = {};
  int col = lane & 15;
  if (col < 2) {
    int kb = (lane < 16) ? 0 : 8;
#pragma unroll
    for (int j = 0; j < 16; ++j) b[j] = (_Float16)W2[(kb + j + ((j >= 8) ? 8 : 0)) * 2 + col];
  }
  v8f c = {};
  c = __builtin_amdgcn_wmma_f32_16x16x32_f16(false, a, false, b, (short)0, c, false, false);

  if (col < 2) {
    int mofs = (lane < 16) ? 0 : 8;
#pragma unroll
    for (int r = 0; r < 8; ++r) {
      int n2 = base + r + mofs;
      if (n2 < N) t2[n2 * 2 + col] = c[r];
    }
  }
}

__global__ void k_logits2(const float* __restrict__ t2,
                          const float* __restrict__ as2,
                          const float* __restrict__ ad2,
                          float* __restrict__ als2, float* __restrict__ ald2,
                          unsigned* __restrict__ m2, float* __restrict__ den2,
                          float* __restrict__ acc2, int N) {
  int i = blockIdx.x * blockDim.x + threadIdx.x;
  if (i >= N) return;
  float2 v = ((const float2*)t2)[i];
  float s = v.x * as2[0] + v.y * as2[1];
  float d = v.x * ad2[0] + v.y * ad2[1];
  als2[i] = s; ald2[i] = d;
  m2[i] = fenc(lrelu(s + d));
  den2[i] = 0.f;
  ((float2*)acc2)[i] = make_float2(0.f, 0.f);
}

__global__ void k_edge_max2(const int* __restrict__ ei,
                            const float* __restrict__ als2,
                            const float* __restrict__ ald2,
                            unsigned* __restrict__ m2, int E) {
  int e = blockIdx.x * blockDim.x + threadIdx.x;
  if (e >= E) return;
  int s = ei[e], d = ei[E + e];
  atomicMax(&m2[d], fenc(lrelu(als2[s] + ald2[d])));
}

__global__ void k_edge_acc2(const int* __restrict__ ei,
                            const float* __restrict__ als2,
                            const float* __restrict__ ald2,
                            const unsigned* __restrict__ m2,
                            const float* __restrict__ t2,
                            float* __restrict__ den2, float* __restrict__ acc2,
                            int E) {
  int e = blockIdx.x * blockDim.x + threadIdx.x;
  if (e >= E) return;
  int s = ei[e], d = ei[E + e];
  float w = __expf(lrelu(als2[s] + ald2[d]) - fdec(m2[d]));
  float2 v = ((const float2*)t2)[s];
  atomicAdd(&den2[d], w);
  atomicAdd(&acc2[d * 2 + 0], w * v.x);
  atomicAdd(&acc2[d * 2 + 1], w * v.y);
}

__global__ void k_finalize2(const float* __restrict__ t2,
                            const float* __restrict__ als2,
                            const float* __restrict__ ald2,
                            const unsigned* __restrict__ m2,
                            const float* __restrict__ den2,
                            const float* __restrict__ acc2,
                            const float* __restrict__ b2,
                            float* __restrict__ out, int N) {
  int i = blockIdx.x * blockDim.x + threadIdx.x;
  if (i >= N) return;
  float ws = __expf(lrelu(als2[i] + ald2[i]) - fdec(m2[i]));
  float rden = 1.f / (den2[i] + ws + 1e-16f);
  float2 tv = ((const float2*)t2)[i];
  float2 av = ((const float2*)acc2)[i];
  float2 o;
  o.x = (av.x + ws * tv.x) * rden + b2[0];
  o.y = (av.y + ws * tv.y) * rden + b2[1];
  ((float2*)out)[i] = o;
}

extern "C" void kernel_launch(void* const* d_in, const int* in_sizes, int n_in,
                              void* d_out, int out_size, void* d_ws, size_t ws_size,
                              hipStream_t stream) {
  const float* x   = (const float*)d_in[0];
  const int*   ei  = (const int*)d_in[1];     // [2,E] (src row then dst row)
  const float* W1  = (const float*)d_in[3];
  const float* as1 = (const float*)d_in[4];
  const float* ad1 = (const float*)d_in[5];
  const float* b1  = (const float*)d_in[6];
  const float* W2  = (const float*)d_in[7];
  const float* as2 = (const float*)d_in[8];
  const float* ad2 = (const float*)d_in[9];
  const float* b2  = (const float*)d_in[10];
  float* out = (float*)d_out;

  const int N = in_sizes[0] / 2;
  const int E = in_sizes[1] / 2;

  // Workspace carve-up (~35 MB total, fully L2-resident on MI455X)
  float*    h1   = (float*)d_ws;                         // N*32 (h1, then h2 in place)
  float*    als1 = h1   + (size_t)N * 32;                // N*4
  float*    ald1 = als1 + (size_t)N * 4;                 // N*4
  unsigned* m1   = (unsigned*)(ald1 + (size_t)N * 4);    // N*4
  float*    den1 = (float*)(m1 + (size_t)N * 4);         // N*4
  float*    acc1 = den1 + (size_t)N * 4;                 // N*32
  float*    t2   = acc1 + (size_t)N * 32;                // N*2
  float*    als2 = t2   + (size_t)N * 2;                 // N
  float*    ald2 = als2 + (size_t)N;                     // N
  unsigned* m2   = (unsigned*)(ald2 + (size_t)N);        // N
  float*    den2 = (float*)(m2 + (size_t)N);             // N
  float*    acc2 = den2 + (size_t)N;                     // N*2

  const dim3 blk(256);
  const int tiles   = (N + 15) / 16;
  const int tblocks = (tiles + 7) / 8;          // 8 waves per 256-thread block
  const int nblocks  = (N + 255) / 256;
  const int n4blocks = (N * 4 + 255) / 256;
  const int eblocks  = (E + 255) / 256;

  // Layer 1
  k_transform1<<<tblocks, blk, 0, stream>>>(x, W1, h1, N);
  k_logits1<<<nblocks, blk, 0, stream>>>(h1, as1, ad1, als1, ald1, m1, den1, acc1, N);
  k_edge_max1<<<eblocks, blk, 0, stream>>>(ei, (const float4*)als1, (const float4*)ald1, m1, E);
  k_edge_acc1<<<eblocks, blk, 0, stream>>>(ei, (const float4*)als1, (const float4*)ald1,
                                           (const uint4*)m1, h1, den1, acc1, E);
  k_finalize1<<<n4blocks, blk, 0, stream>>>(h1, als1, ald1, m1, den1, acc1, b1, N);

  // Layer 2 (h2 lives in h1's buffer)
  k_transform2<<<tblocks, blk, 0, stream>>>(h1, W2, t2, N);
  k_logits2<<<nblocks, blk, 0, stream>>>(t2, as2, ad2, als2, ald2, m2, den2, acc2, N);
  k_edge_max2<<<eblocks, blk, 0, stream>>>(ei, als2, ald2, m2, E);
  k_edge_acc2<<<eblocks, blk, 0, stream>>>(ei, als2, ald2, m2, t2, den2, acc2, E);
  k_finalize2<<<nblocks, blk, 0, stream>>>(t2, als2, ald2, m2, den2, acc2, b2, out, N);
}
